// AEVPotential_56143812494149
// MI455X (gfx1250) — matrix operand order repaired and checked by reading, not compiled, not run on hardware
//
#include <hip/hip_runtime.h>
#include <hip/hip_bf16.h>

typedef __attribute__((ext_vector_type(16))) _Float16 v16h;
typedef __attribute__((ext_vector_type(8)))  _Float16 v8h;
typedef __attribute__((ext_vector_type(8)))  float    v8f;

#define NC   128   // conformations
#define NA   64    // atoms
#define NK   24    // neighbors
#define NS   4     // species
#define NT_PAIRS 276  // K*(K-1)/2
#define NFEAT 384

#define PI_F 3.14159265358979f

// per-species repacked weight region (f16 elements)
#define W1_ELEMS (384*160)   // 61440, KT=12 NT=10
#define W2_ELEMS (160*128)   // 20480, KT=5  NT=8
#define W3_ELEMS (128*96)    // 12288, KT=4  NT=6
#define WSP_ELEMS (W1_ELEMS + W2_ELEMS + W3_ELEMS)  // 94208

__device__ __constant__ int d_triu[4][4] = {
    {0,1,2,3},{1,4,5,6},{2,5,7,8},{3,6,8,9}};

__device__ __forceinline__ float celu01(float x) {
    return x > 0.f ? x : 0.1f * (__expf(x * 10.f) - 1.f);
}

__device__ __forceinline__ float fcut(float d, float rc) {
    return d < rc ? 0.5f * __cosf(PI_F * d / rc) + 0.5f : 0.f;
}

// ---------------- kernel 0: zero output ----------------
__global__ void zero_out_kernel(float* out) {
    int t = threadIdx.x;
    if (t < NC) out[t] = 0.f;
}

// ---------------- kernel 1: AEV construction ----------------
// block = 128 threads = 4 waves = 4 atoms; grid = (C*A)/4
__global__ __launch_bounds__(128) void aev_kernel(
                           const int* __restrict__ elem,
                           const int* __restrict__ nbr,
                           const float* __restrict__ dist,
                           const float* __restrict__ diff,
                           _Float16* __restrict__ aev) {
    __shared__ float s_aev[4][NFEAT];
    const int w    = threadIdx.x >> 5;
    const int lane = threadIdx.x & 31;
    const int atom = blockIdx.x * 4 + w;
    const int c    = atom >> 6;        // / NA

    for (int e = threadIdx.x; e < 4 * NFEAT; e += blockDim.x)
        ((float*)s_aev)[e] = 0.f;
    __syncthreads();

    // ---- radial: one lane per neighbor ----
    if (lane < NK) {
        const int   k  = lane;
        const int   nb = nbr[atom * NK + k];
        const int   ns = elem[c * NA + nb];
        const float d  = dist[atom * NK + k];
        const float fr = fcut(d, 5.2f);
        #pragma unroll
        for (int r = 0; r < 16; ++r) {
            float sh = 0.9f + 0.26875f * (float)r;
            float dr = d - sh;
            float v  = 0.25f * __expf(-16.f * dr * dr) * fr;
            atomicAdd(&s_aev[w][ns * 16 + r], v);
        }
    }

    // ---- angular: 276 pairs over 32 lanes ----
    for (int it = 0; it < 9; ++it) {
        int t = it * 32 + lane;
        if (t < NT_PAIRS) {
            int j = 0, t2 = t;
            while (t2 >= NK - 1 - j) { t2 -= NK - 1 - j; ++j; }
            int k = j + 1 + t2;

            const float d1 = dist[atom * NK + j];
            const float d2 = dist[atom * NK + k];
            const float* v1 = diff + (size_t)(atom * NK + j) * 3;
            const float* v2 = diff + (size_t)(atom * NK + k) * 3;
            float dot = v1[0]*v2[0] + v1[1]*v2[1] + v1[2]*v2[2];
            float cosang = dot / (d1 * d2);
            float ang = acosf(0.95f * cosang);
            float fc12 = fcut(d1, 3.5f) * fcut(d2, 3.5f);
            float davg = 0.5f * (d1 + d2);

            float f2[4];
            #pragma unroll
            for (int q = 0; q < 4; ++q) {
                float sh = 0.9f + 0.65f * (float)q;
                float dd = davg - sh;
                f2[q] = __expf(-8.f * dd * dd);
            }

            const int s1 = elem[c * NA + nbr[atom * NK + j]];
            const int s2 = elem[c * NA + nbr[atom * NK + k]];
            const int p  = d_triu[s1][s2];
            float* dst = &s_aev[w][64 + p * 32];

            #pragma unroll
            for (int z = 0; z < 8; ++z) {
                float shz = ((float)z + 0.5f) * (PI_F / 8.f);
                float base = 0.5f * (1.f + __cosf(ang - shz));
                float p2 = base * base;
                float p4 = p2 * p2;
                float p8 = p4 * p4;
                float p16 = p8 * p8;
                float f1 = p16 * p16;                 // ^32
                float common = 2.f * f1 * fc12;
                #pragma unroll
                for (int q = 0; q < 4; ++q)
                    atomicAdd(&dst[z * 4 + q], common * f2[q]);
            }
        }
    }
    __syncthreads();

    _Float16* row = aev + (size_t)atom * NFEAT;
    for (int e = lane; e < NFEAT; e += 32)
        row[e] = (_Float16)s_aev[w][e];
}

// ---------------- kernel 2: weight repack f32 -> f16 B-fragments ----
// B fragment (32x16 KxN), k order chosen here (A loader uses the same): lane l: N = l&15, elems e: K = (l>>4)*16 + e
// fragment (kt,nt) stored contiguously: frag*512 + lane*16 + e
__global__ void repack_kernel(const float* __restrict__ W1,
                              const float* __restrict__ W2,
                              const float* __restrict__ W3,
                              _Float16* __restrict__ wrep) {
    int idx = blockIdx.x * blockDim.x + threadIdx.x;
    if (idx >= NS * WSP_ELEMS) return;
    int s = idx / WSP_ELEMS;
    int r = idx - s * WSP_ELEMS;

    const float* W; int FOUT, NT, off;
    if (r < W1_ELEMS)                { W = W1; FOUT = 160; NT = 10; off = r;            }
    else if (r < W1_ELEMS + W2_ELEMS){ W = W2; FOUT = 128; NT = 8;  off = r - W1_ELEMS; }
    else                             { W = W3; FOUT = 96;  NT = 6;  off = r - W1_ELEMS - W2_ELEMS; }

    int frag   = off >> 9;
    int within = off & 511;
    int lane   = within >> 4;
    int e      = within & 15;
    int kt     = frag / NT;
    int nt     = frag - kt * NT;
    int fin    = kt * 32 + (lane >> 4) * 16 + e;
    int fout   = nt * 16 + (lane & 15);
    int FIN    = (FOUT == 160) ? 384 : (FOUT == 128 ? 160 : 128);

    wrep[idx] = (_Float16)W[(size_t)s * FIN * FOUT + (size_t)fin * FOUT + fout];
}

// ---------------- WMMA fragment loaders ----------------
// A fragment 16x32 f16, k order chosen here (matches B repack):
//   lane l: row M = l&15; elems e: K = (l>>4)*16 + e
__device__ __forceinline__ v16h load_afrag_h(const _Float16* t, int ld, int kt, int lane) {
    int row = lane & 15, hi = lane >> 4;
    const _Float16* p = t + row * ld + kt * 32 + hi * 16;
    v8h lo = *(const v8h*)p;
    v8h hh = *(const v8h*)(p + 8);
    return __builtin_shufflevector(lo, hh, 0,1,2,3,4,5,6,7,8,9,10,11,12,13,14,15);
}

__device__ __forceinline__ v16h load_bfrag(const _Float16* fragbase, int lane) {
    const _Float16* p = fragbase + lane * 16;
    v8h x = *(const v8h*)p;
    v8h y = *(const v8h*)(p + 8);
    return __builtin_shufflevector(x, y, 0,1,2,3,4,5,6,7,8,9,10,11,12,13,14,15);
}

// ---------------- one WMMA layer: f16 LDS input -> f16 LDS output ------
// B fragments double-buffered so the next kt's global load issues before
// the current WMMA consumes its operand.
template<int KT, int NTL, int LDIN, int LDOUT>
__device__ __forceinline__ void layer_f16(const _Float16* in, const _Float16* wb,
                                          const float* bias, _Float16* outb, int lane) {
    const int col = lane & 15, hi = lane >> 4;
    for (int nt = 0; nt < NTL; ++nt) {
        v16h bcur = load_bfrag(wb + (size_t)nt * 512, lane);
        v8f acc = {};
        #pragma unroll
        for (int kt = 0; kt < KT; ++kt) {
            v16h bn = bcur;
            if (kt + 1 < KT)
                bn = load_bfrag(wb + (size_t)((kt + 1) * NTL + nt) * 512, lane);
            v16h a = load_afrag_h(in, LDIN, kt, lane);
            acc = __builtin_amdgcn_wmma_f32_16x16x32_f16(false, a, false, bcur,
                                                         (short)0, acc, false, false);
            bcur = bn;
        }
        float bv = bias[nt * 16 + col];
        #pragma unroll
        for (int v = 0; v < 8; ++v)
            outb[(v + 8 * hi) * LDOUT + nt * 16 + col] = (_Float16)celu01(acc[v] + bv);
    }
}

// ---------------- kernel 3: MLP, 4 waves per block (wave = species) ----
__global__ __launch_bounds__(128, 1) void mlp_kernel(
                           const _Float16* __restrict__ aev,
                           const _Float16* __restrict__ wrep,
                           const float* __restrict__ b1,
                           const float* __restrict__ b2,
                           const float* __restrict__ b3,
                           const float* __restrict__ W4,
                           const float* __restrict__ b4,
                           const int*   __restrict__ elem,
                           float* __restrict__ out) {
    __shared__ __align__(16) _Float16 s_tile[16 * NFEAT];    // 12 KB AEV tile
    __shared__ __align__(16) _Float16 s_hA[NS][16 * 160];    // per-wave ping
    __shared__ __align__(16) _Float16 s_hB[NS][16 * 160];    // per-wave pong

    const int lane = threadIdx.x & 31;
    const int s    = threadIdx.x >> 5;           // wave id == species
    const int a0   = blockIdx.x * 16;            // first flattened atom of tile
    const int conf = a0 / NA;
    const int col = lane & 15;

    // ---- stage AEV tile into LDS with async global->LDS copies ----
    {
        const char* gsrc = (const char*)(aev + (size_t)a0 * NFEAT);
        #pragma unroll
        for (int it = 0; it < 6; ++it) {         // 16*384*2 B = 768 x 16B chunks
            int off = (threadIdx.x + it * 128) * 16;
            unsigned int lds_off = (unsigned int)(size_t)((char*)s_tile + off);
            unsigned long long ga = (unsigned long long)(size_t)(gsrc + off);
            asm volatile("global_load_async_to_lds_b128 %0, %1, off"
                         :: "v"(lds_off), "v"(ga) : "memory");
        }
        asm volatile("s_wait_asynccnt 0x0" ::: "memory");
    }
    __syncthreads();

    int myspec = -1;
    if (lane < 16) myspec = elem[a0 + lane];

    const _Float16* wsp = wrep + (size_t)s * WSP_ELEMS;
    const _Float16* w1 = wsp;
    const _Float16* w2 = wsp + W1_ELEMS;
    const _Float16* w3 = wsp + W1_ELEMS + W2_ELEMS;
    __builtin_prefetch(w1, 0, 0);

    // ---- layer 1: 384 -> 160 ----
    layer_f16<12, 10, NFEAT, 160>(s_tile, w1, b1 + s * 160, s_hA[s], lane);
    __builtin_prefetch(w2, 0, 0);
    __syncthreads();

    // ---- layer 2: 160 -> 128 ----
    layer_f16<5, 8, 160, 128>(s_hA[s], w2, b2 + s * 128, s_hB[s], lane);
    __builtin_prefetch(w3, 0, 0);
    __syncthreads();

    // ---- layer 3: 128 -> 96 ----
    layer_f16<4, 6, 128, 96>(s_hB[s], w3, b3 + s * 96, s_hA[s], lane);
    __syncthreads();

    // ---- layer 4: 96 -> 1 (VALU dot), species select ----
    float part = 0.f;
    if (lane < 16) {
        const float* w4 = W4 + s * 96;
        const _Float16* h3 = s_hA[s] + lane * 96;
        float acc4 = b4[s];
        #pragma unroll 8
        for (int f = 0; f < 96; ++f)
            acc4 += (float)h3[f] * w4[f];
        if (myspec == s) part = acc4;
    }

    // wave reduction (lanes 16..31 hold 0); one atomic per wave
    #pragma unroll
    for (int m = 16; m >= 1; m >>= 1)
        part += __shfl_xor(part, m, 32);
    if (lane == 0)
        atomicAdd(&out[conf], part);
}

// ---------------- host launcher ----------------
extern "C" void kernel_launch(void* const* d_in, const int* in_sizes, int n_in,
                              void* d_out, int out_size, void* d_ws, size_t ws_size,
                              hipStream_t stream) {
    const int*   elem = (const int*)  d_in[0];   // (C,A)
    const int*   nbr  = (const int*)  d_in[1];   // (C,A,K)
    const float* dist = (const float*)d_in[2];   // (C,A,K)
    const float* diff = (const float*)d_in[3];   // (C,A,K,3)
    const float* W1   = (const float*)d_in[4];
    const float* b1   = (const float*)d_in[5];
    const float* W2   = (const float*)d_in[6];
    const float* b2   = (const float*)d_in[7];
    const float* W3   = (const float*)d_in[8];
    const float* b3   = (const float*)d_in[9];
    const float* W4   = (const float*)d_in[10];
    const float* b4   = (const float*)d_in[11];
    float* out = (float*)d_out;

    _Float16* aev  = (_Float16*)d_ws;                       // C*A*384 f16
    _Float16* wrep = aev + (size_t)NC * NA * NFEAT;         // 4*WSP_ELEMS f16

    zero_out_kernel<<<1, 128, 0, stream>>>(out);

    aev_kernel<<<(NC * NA) / 4, 128, 0, stream>>>(elem, nbr, dist, diff, aev);

    int nrep = NS * WSP_ELEMS;
    repack_kernel<<<(nrep + 255) / 256, 256, 0, stream>>>(W1, W2, W3, wrep);

    mlp_kernel<<<(NC * NA) / 16, 128, 0, stream>>>(aev, wrep, b1, b2, b3,
                                                   W4, b4, elem, out);
}
